// MLA_18279380812506
// MI455X (gfx1250) — matrix operand order repaired and checked by reading, not compiled, not run on hardware
//
#include <hip/hip_runtime.h>
#include <cstdint>
#include <cstddef>

// ---------------------------------------------------------------------------
// MLA forward for MI455X (gfx1250): bf16 WMMA everywhere, B128 fragment loads.
// ---------------------------------------------------------------------------

typedef __attribute__((ext_vector_type(16))) __bf16 v16bf;
typedef __attribute__((ext_vector_type(8)))  float  v8f;

#define S_LEN   2048
#define HID_D   2048
#define NH      16
#define QLR_D   1536
#define KVLR_D  512
#define ROPE_D  64
#define NOPE_D  128
#define VD_D    128
#define QKD_D   192
#define DQK     576            // KVLR + ROPE (combined absorbed-Q / K dim)
#define LN10K   9.210340371976184f
#define SM_SCALE 0.07216878364870323f   // 1/sqrt(192)

static __device__ __forceinline__ v8f wmma_bf16(v16bf a, v16bf b, v8f c) {
  return __builtin_amdgcn_wmma_f32_16x16x32_bf16(false, a, false, b, (short)0, c,
                                                 false, false);
}

union Frag {
  v16bf v;
  uint4 q[2];
};

// A-operand fragment: elements 0..15 at p[0..15] (contiguous in K)
static __device__ __forceinline__ v16bf ldA(const __bf16* p) {
  Frag f;
  f.q[0] = *(const uint4*)(p);
  f.q[1] = *(const uint4*)(p + 8);
  return f.v;
}
// B-operand fragment: elements 0..15 at p[0..15] (contiguous in K)
static __device__ __forceinline__ v16bf ldB(const __bf16* p) {
  Frag f;
  f.q[0] = *(const uint4*)(p);
  f.q[1] = *(const uint4*)(p + 8);
  return f.v;
}

// f32 -> bf16 copy
__global__ void k_f2bf(const float* __restrict__ in, __bf16* __restrict__ out, int n) {
  int i = blockIdx.x * blockDim.x + threadIdx.x;
  if (i < n) out[i] = (__bf16)in[i];
}

// wkv_b nope-part transposed: out[h*65536 + c*128 + d] = w[(h*256+d)*512 + c]
__global__ void k_wkvb_nope_T(const float* __restrict__ w, __bf16* __restrict__ out) {
  int i = blockIdx.x * blockDim.x + threadIdx.x;   // h*c*d, d fastest
  int d = i & (NOPE_D - 1);
  int c = (i >> 7) & (KVLR_D - 1);
  int h = i >> 16;
  out[i] = (__bf16)w[((size_t)(h * 256 + d)) * KVLR_D + c];
}

// Vt[c*2048 + t] = Kc[t*576 + c]  (c < 512)
__global__ void k_vT(const __bf16* __restrict__ Kc, __bf16* __restrict__ Vt) {
  int i = blockIdx.x * blockDim.x + threadIdx.x;   // c*S + t, t fastest
  int t = i & (S_LEN - 1);
  int c = i >> 11;
  Vt[i] = Kc[(size_t)t * DQK + c];
}

// ---------------------------------------------------------------------------
// Generic batched bf16 GEMM via WMMA 16x16x32, contiguous-K B operand:
//   C(b, m, n) = sum_k A[b*aB + m*lda + k] * B[b*bB + n*bsn + k]  (+ bias[n])
// One wave computes a 32(M) x 64(N) tile; 4 waves/block stack along M (128).
// grid = (N/64, M/128, batch), block = 128.  All dims multiples of 16/32/64.
// ---------------------------------------------------------------------------
__global__ __launch_bounds__(128)
void k_gemm_wmma(const __bf16* __restrict__ A, long lda, long aB,
                 const __bf16* __restrict__ B, long bsn, long bB,
                 const float* __restrict__ bias,
                 float* __restrict__ Cf, __bf16* __restrict__ Cb,
                 long ldc, long cB, int K) {
  const int lane = threadIdx.x & 31;
  const int wave = threadIdx.x >> 5;
  const int l16  = lane & 15;
  const int hi   = lane >> 4;
  const long nBase = (long)blockIdx.x * 64;
  const long mBase = (long)blockIdx.y * 128 + wave * 32;
  const int  batch = blockIdx.z;

  const __bf16* Ab = A + (long)batch * aB;
  const __bf16* Bb = B + (long)batch * bB;

  const __bf16* aRow0 = Ab + (mBase + l16) * lda + hi * 16;
  const __bf16* aRow1 = aRow0 + 16 * lda;

  v8f acc[2][4] = {};

  for (int kk = 0; kk < K; kk += 32) {
    const v16bf a0 = ldA(aRow0 + kk);
    const v16bf a1 = ldA(aRow1 + kk);
#pragma unroll
    for (int t = 0; t < 4; ++t) {
      const __bf16* bp = Bb + (nBase + t * 16 + l16) * bsn + kk + hi * 16;
      const v16bf bfr = ldB(bp);
      acc[0][t] = wmma_bf16(a0, bfr, acc[0][t]);
      acc[1][t] = wmma_bf16(a1, bfr, acc[1][t]);
    }
  }

#pragma unroll
  for (int mi = 0; mi < 2; ++mi) {
#pragma unroll
    for (int t = 0; t < 4; ++t) {
      const long n = nBase + t * 16 + l16;
      const float bv = bias ? bias[n] : 0.0f;
#pragma unroll
      for (int j = 0; j < 8; ++j) {
        const long m = mBase + mi * 16 + j + 8 * hi;
        const float v = acc[mi][t][j] + bv;
        const long off = (long)batch * cB + m * ldc + n;
        if (Cf) Cf[off] = v;
        if (Cb) Cb[off] = (__bf16)v;
      }
    }
  }
}

// ---------------------------------------------------------------------------
// RMSNorm (row per block), fp32 in -> bf16 out
// ---------------------------------------------------------------------------
__global__ void k_rmsnorm(const float* __restrict__ in, const float* __restrict__ w,
                          __bf16* __restrict__ out, int dim) {
  const long rowOff = (long)blockIdx.x * dim;
  __shared__ float red[8];
  float ss = 0.f;
  for (int i = threadIdx.x; i < dim; i += blockDim.x) {
    float v = in[rowOff + i];
    ss += v * v;
  }
  for (int o = 16; o >= 1; o >>= 1) ss += __shfl_xor(ss, o, 32);
  if ((threadIdx.x & 31) == 0) red[threadIdx.x >> 5] = ss;
  __syncthreads();
  float tot = 0.f;
  const int nw = blockDim.x >> 5;
  for (int i = 0; i < nw; ++i) tot += red[i];
  const float inv = rsqrtf(tot / (float)dim + 1e-6f);
  for (int i = threadIdx.x; i < dim; i += blockDim.x)
    out[rowOff + i] = (__bf16)(in[rowOff + i] * inv * w[i]);
}

// ---------------------------------------------------------------------------
// kv_full [S,576] fp32 -> Kc [S,576] bf16: rmsnorm first 512, rope last 64
// ---------------------------------------------------------------------------
__global__ void k_kv_finalize(const float* __restrict__ kvf,
                              const float* __restrict__ kvw,
                              __bf16* __restrict__ Kc) {
  const int s = blockIdx.x;
  const float* rowp = kvf + (long)s * DQK;
  __shared__ float red[8];
  float ss = 0.f;
  for (int i = threadIdx.x; i < KVLR_D; i += blockDim.x) {
    float v = rowp[i];
    ss += v * v;
  }
  for (int o = 16; o >= 1; o >>= 1) ss += __shfl_xor(ss, o, 32);
  if ((threadIdx.x & 31) == 0) red[threadIdx.x >> 5] = ss;
  __syncthreads();
  float tot = 0.f;
  const int nw = blockDim.x >> 5;
  for (int i = 0; i < nw; ++i) tot += red[i];
  const float inv = rsqrtf(tot / (float)KVLR_D + 1e-6f);
  for (int i = threadIdx.x; i < KVLR_D; i += blockDim.x)
    Kc[(long)s * DQK + i] = (__bf16)(rowp[i] * inv * kvw[i]);
  // rope k_pe
  for (int r = threadIdx.x; r < ROPE_D; r += blockDim.x) {
    const int j = (r < 32) ? r : r - 32;
    const float invf = __expf(-(2.0f * (float)j / 64.0f) * LN10K);
    const float ang = (float)s * invf;
    const float c = __cosf(ang), sn = __sinf(ang);
    const float x0 = rowp[KVLR_D + j];
    const float x1 = rowp[KVLR_D + j + 32];
    const float v = (r < 32) ? (x0 * c - x1 * sn) : (x1 * c + x0 * sn);
    Kc[(long)s * DQK + KVLR_D + r] = (__bf16)v;
  }
}

// ---------------------------------------------------------------------------
// qb [S,3072] fp32 -> qnope bf16 [S, H*128], rope(q_pe) into Qc[...,512:576]
// ---------------------------------------------------------------------------
__global__ void k_q_extract(const float* __restrict__ qb,
                            __bf16* __restrict__ qnope,
                            __bf16* __restrict__ Qc) {
  const int s = blockIdx.x;
  const float* rowp = qb + (long)s * (NH * QKD_D);
  for (int idx = threadIdx.x; idx < NH * QKD_D; idx += blockDim.x) {
    const int h = idx / QKD_D;
    const int d = idx % QKD_D;
    if (d < NOPE_D) {
      qnope[(long)s * (NH * NOPE_D) + h * NOPE_D + d] = (__bf16)rowp[idx];
    } else {
      const int r = d - NOPE_D;
      const int j = (r < 32) ? r : r - 32;
      const float invf = __expf(-(2.0f * (float)j / 64.0f) * LN10K);
      const float ang = (float)s * invf;
      const float c = __cosf(ang), sn = __sinf(ang);
      const float x0 = rowp[h * QKD_D + NOPE_D + j];
      const float x1 = rowp[h * QKD_D + NOPE_D + j + 32];
      const float v = (r < 32) ? (x0 * c - x1 * sn) : (x1 * c + x0 * sn);
      Qc[(long)s * (NH * DQK) + h * DQK + KVLR_D + r] = (__bf16)v;
    }
  }
}

// ---------------------------------------------------------------------------
// Flash attention over the absorbed latent space.
//   Qc : [S, H, 576] bf16   (absorbed q_nope(512) | roped q_pe(64))
//   Kc : [S, 576]    bf16   (rmsnorm kv(512)      | roped k_pe(64))
//   Vt : [512, S]    bf16   (transposed latent V for contiguous-K PV loads)
//   O  : [S, H, 512] bf16
// Block = (16-query tile, head); 4 waves, each owns a 128-wide V chunk.
// ---------------------------------------------------------------------------
__global__ __launch_bounds__(128)
void k_mla_attn(const __bf16* __restrict__ Qc, const __bf16* __restrict__ Kc,
                const __bf16* __restrict__ Vt, __bf16* __restrict__ O) {
  const int lane = threadIdx.x & 31;
  const int wave = threadIdx.x >> 5;
  const int l16  = lane & 15;
  const int hi   = lane >> 4;
  const int s0   = blockIdx.x * 16;
  const int h    = blockIdx.y;
  const int vbase = wave * 128;

  __shared__ __bf16 pShared[4][16 * 32];
  __bf16* myP = pShared[wave];

  // Cache Q fragments: 18 chunks of K=32 over 576 (two B128 loads each)
  v16bf qf[18];
  {
    const __bf16* qptr = Qc + (long)(s0 + l16) * (NH * DQK) + (long)h * DQK + hi * 16;
#pragma unroll
    for (int c = 0; c < 18; ++c) qf[c] = ldA(qptr + c * 32);
  }

  float m[8], l[8];
#pragma unroll
  for (int j = 0; j < 8; ++j) { m[j] = -1e30f; l[j] = 0.f; }
  v8f oacc[8] = {};

  const int nT = ((s0 + 15) >> 5) + 1;   // 32-key blocks (causal)
  for (int tb = 0; tb < nT; ++tb) {
    const int t0 = tb * 32;
    // ---- scores: two 16x16 tiles over 576-dim contraction ----
    v8f sc[2] = {};
#pragma unroll
    for (int c = 0; c < 18; ++c) {
#pragma unroll
      for (int tt = 0; tt < 2; ++tt) {
        const __bf16* kp = Kc + (long)(t0 + tt * 16 + l16) * DQK + c * 32 + hi * 16;
        sc[tt] = wmma_bf16(qf[c], ldB(kp), sc[tt]);
      }
    }
    // ---- scale + causal mask + online softmax ----
    float sval[2][8];
#pragma unroll
    for (int tt = 0; tt < 2; ++tt)
#pragma unroll
      for (int j = 0; j < 8; ++j) {
        const int t = t0 + tt * 16 + l16;
        const int sRow = s0 + j + 8 * hi;
        float v = sc[tt][j] * SM_SCALE;
        if (t > sRow) v = -1e30f;
        sval[tt][j] = v;
      }
#pragma unroll
    for (int j = 0; j < 8; ++j) {
      float mx = fmaxf(sval[0][j], sval[1][j]);
      for (int o = 8; o >= 1; o >>= 1) mx = fmaxf(mx, __shfl_xor(mx, o, 32));
      const float mnew = fmaxf(m[j], mx);
      const float corr = __expf(m[j] - mnew);
      m[j] = mnew;
      l[j] *= corr;
#pragma unroll
      for (int vt = 0; vt < 8; ++vt) oacc[vt][j] *= corr;
      const float p0 = __expf(sval[0][j] - mnew);
      const float p1 = __expf(sval[1][j] - mnew);
      float ps = p0 + p1;
      for (int o = 8; o >= 1; o >>= 1) ps += __shfl_xor(ps, o, 32);
      l[j] += ps;
      sval[0][j] = p0;
      sval[1][j] = p1;
    }
    // ---- stage P through LDS: C-layout -> A-layout ----
#pragma unroll
    for (int tt = 0; tt < 2; ++tt)
#pragma unroll
      for (int j = 0; j < 8; ++j)
        myP[(j + 8 * hi) * 32 + tt * 16 + l16] = (__bf16)sval[tt][j];
    __syncthreads();
    v16bf pf;
    {
      Frag f;
      const __bf16* pp = myP + l16 * 32 + hi * 16;
      f.q[0] = *(const uint4*)(pp);
      f.q[1] = *(const uint4*)(pp + 8);
      pf = f.v;
    }
    __syncthreads();
    // ---- P @ V  via transposed V (contiguous in K=t) ----
#pragma unroll
    for (int vt = 0; vt < 8; ++vt) {
      const int col = vbase + vt * 16 + l16;
      const __bf16* vp = Vt + (long)col * S_LEN + t0 + hi * 16;
      oacc[vt] = wmma_bf16(pf, ldB(vp), oacc[vt]);
    }
  }

  // ---- normalize + store ----
#pragma unroll
  for (int vt = 0; vt < 8; ++vt) {
    const int col = vbase + vt * 16 + l16;
#pragma unroll
    for (int j = 0; j < 8; ++j) {
      const long s = s0 + j + 8 * hi;
      const float v = oacc[vt][j] / l[j];
      O[s * (NH * KVLR_D) + h * KVLR_D + col] = (__bf16)v;
    }
  }
}

// ---------------------------------------------------------------------------
// Host-side orchestration
// ---------------------------------------------------------------------------
extern "C" void kernel_launch(void* const* d_in, const int* in_sizes, int n_in,
                              void* d_out, int out_size, void* d_ws, size_t ws_size,
                              hipStream_t stream) {
  (void)in_sizes; (void)n_in; (void)out_size; (void)ws_size;
  const float* x        = (const float*)d_in[0];
  const float* wq_a_w   = (const float*)d_in[1];
  const float* wq_a_b   = (const float*)d_in[2];
  const float* q_norm_w = (const float*)d_in[3];
  const float* wq_b_w   = (const float*)d_in[4];
  const float* wq_b_b   = (const float*)d_in[5];
  const float* wkv_a_w  = (const float*)d_in[6];
  const float* wkv_a_b  = (const float*)d_in[7];
  const float* kv_norm_w= (const float*)d_in[8];
  const float* wkv_b_w  = (const float*)d_in[9];
  const float* wo_w     = (const float*)d_in[10];
  const float* wo_b     = (const float*)d_in[11];
  float* out            = (float*)d_out;

  char* ws = (char*)d_ws;
  size_t off = 0;
  auto take = [&](size_t bytes) -> char* {
    char* p = ws + off;
    off = (off + bytes + 255) & ~(size_t)255;
    return p;
  };
  const size_t nx    = (size_t)S_LEN * HID_D;
  const size_t nwqa  = (size_t)QLR_D * HID_D;
  const size_t nwqb  = (size_t)(NH * QKD_D) * QLR_D;
  const size_t nwkva = (size_t)DQK * HID_D;
  const size_t nwkvb = (size_t)(NH * 256) * KVLR_D;
  const size_t nwo   = (size_t)HID_D * (NH * VD_D);
  const size_t nwkvbT = (size_t)NH * KVLR_D * NOPE_D;
  const size_t nvT    = (size_t)KVLR_D * S_LEN;

  __bf16* xb     = (__bf16*)take(nx * 2);
  __bf16* wqab   = (__bf16*)take(nwqa * 2);
  __bf16* wqbb   = (__bf16*)take(nwqb * 2);
  __bf16* wkvab  = (__bf16*)take(nwkva * 2);
  __bf16* wkvbb  = (__bf16*)take(nwkvb * 2);
  __bf16* wkvbT  = (__bf16*)take(nwkvbT * 2);
  __bf16* wob    = (__bf16*)take(nwo * 2);
  float*  qa_f   = (float*)take((size_t)S_LEN * QLR_D * 4);
  __bf16* qn_b   = (__bf16*)take((size_t)S_LEN * QLR_D * 2);
  float*  qb_f   = (float*)take((size_t)S_LEN * NH * QKD_D * 4);
  float*  kvf_f  = (float*)take((size_t)S_LEN * DQK * 4);
  __bf16* qnope  = (__bf16*)take((size_t)S_LEN * NH * NOPE_D * 2);
  __bf16* Qc     = (__bf16*)take((size_t)S_LEN * NH * DQK * 2);
  __bf16* Kc     = (__bf16*)take((size_t)S_LEN * DQK * 2);
  __bf16* Vt     = (__bf16*)take(nvT * 2);
  __bf16* attnb  = (__bf16*)take((size_t)S_LEN * NH * KVLR_D * 2);
  __bf16* oin    = (__bf16*)take((size_t)S_LEN * (NH * VD_D) * 2);

  auto cvt = [&](const float* src, __bf16* dst, size_t n) {
    k_f2bf<<<dim3((unsigned)((n + 255) / 256)), dim3(256), 0, stream>>>(src, dst, (int)n);
  };
  cvt(x, xb, nx);
  cvt(wq_a_w, wqab, nwqa);
  cvt(wq_b_w, wqbb, nwqb);
  cvt(wkv_a_w, wkvab, nwkva);
  cvt(wkv_b_w, wkvbb, nwkvb);
  cvt(wo_w, wob, nwo);
  k_wkvb_nope_T<<<dim3((unsigned)(nwkvbT / 256)), dim3(256), 0, stream>>>(wkv_b_w, wkvbT);

  // 1) q_a = x @ wq_a^T + b      [2048,1536]
  k_gemm_wmma<<<dim3(QLR_D / 64, S_LEN / 128, 1), dim3(128), 0, stream>>>(
      xb, HID_D, 0, wqab, HID_D, 0, wq_a_b, qa_f, nullptr, QLR_D, 0, HID_D);
  // 2) rmsnorm(q_a) -> bf16
  k_rmsnorm<<<dim3(S_LEN), dim3(256), 0, stream>>>(qa_f, q_norm_w, qn_b, QLR_D);
  // 3) q_b = qn @ wq_b^T + b     [2048,3072]
  k_gemm_wmma<<<dim3((NH * QKD_D) / 64, S_LEN / 128, 1), dim3(128), 0, stream>>>(
      qn_b, QLR_D, 0, wqbb, QLR_D, 0, wq_b_b, qb_f, nullptr, NH * QKD_D, 0, QLR_D);
  // 4) kv_full = x @ wkv_a^T + b [2048,576]
  k_gemm_wmma<<<dim3(DQK / 64, S_LEN / 128, 1), dim3(128), 0, stream>>>(
      xb, HID_D, 0, wkvab, HID_D, 0, wkv_a_b, kvf_f, nullptr, DQK, 0, HID_D);
  // 5) Kc = [rmsnorm(kv) | rope(k_pe)], then Vt = Kc[:,0:512]^T
  k_kv_finalize<<<dim3(S_LEN), dim3(256), 0, stream>>>(kvf_f, kv_norm_w, Kc);
  k_vT<<<dim3((unsigned)(nvT / 256)), dim3(256), 0, stream>>>(Kc, Vt);
  // 6) split q_b -> qnope bf16 + roped q_pe into Qc[...,512:576]
  k_q_extract<<<dim3(S_LEN), dim3(256), 0, stream>>>(qb_f, qnope, Qc);
  // 7) absorb: Qc[...,0:512] = qnope @ wkv_b[h,0:128,:]  via transposed weight
  k_gemm_wmma<<<dim3(KVLR_D / 64, S_LEN / 128, NH), dim3(128), 0, stream>>>(
      qnope, NH * NOPE_D, NOPE_D,
      wkvbT, NOPE_D, (long)KVLR_D * NOPE_D,
      nullptr, nullptr, Qc, NH * DQK, DQK, NOPE_D);
  // 8) flash attention in latent space
  k_mla_attn<<<dim3(S_LEN / 16, NH), dim3(128), 0, stream>>>(Qc, Kc, Vt, attnb);
  // 9) per-head down-proj: oin[:, h*128:+128] = attn_h @ wkv_b[h,128:256,:]^T
  k_gemm_wmma<<<dim3(VD_D / 64, S_LEN / 128, NH), dim3(128), 0, stream>>>(
      attnb, NH * KVLR_D, KVLR_D,
      wkvbb + (size_t)NOPE_D * KVLR_D, KVLR_D, (long)256 * KVLR_D,
      nullptr, nullptr, oin, NH * VD_D, VD_D, KVLR_D);
  // 10) out = oin @ wo^T + wo_b  [2048,2048] fp32 -> d_out
  k_gemm_wmma<<<dim3(HID_D / 64, S_LEN / 128, 1), dim3(128), 0, stream>>>(
      oin, NH * VD_D, 0, wob, NH * VD_D, 0, wo_b, out, nullptr, HID_D, 0, NH * VD_D);
}
